// ImageTransformerDenoiserModel_83219286327373
// MI455X (gfx1250) — compile-verified, matrix operand order checked by reading, not run on hardware
//
#include <hip/hip_runtime.h>
#include <cmath>

typedef unsigned short u16;
typedef unsigned int   u32;

typedef __bf16 bf16_t;
typedef bf16_t v16bf __attribute__((ext_vector_type(16)));
typedef float  v8f   __attribute__((ext_vector_type(8)));

// ---------------------------------------------------------------- helpers

__device__ __forceinline__ u16 f2bf(float f) {
  u32 u = __float_as_uint(f);
  u32 r = (u + 0x7fffu + ((u >> 16) & 1u)) >> 16;   // round-to-nearest-even
  return (u16)r;
}

// Load a 16x32 bf16 fragment (A layout; also valid for B when the source is
// the N x K weight matrix, i.e. C = A * W^T) from a row-major bf16 matrix.
__device__ __forceinline__ v16bf load_frag_rm(const u16* __restrict__ base,
                                              int ld, int row0, int k0) {
  int lane = threadIdx.x & 31;
  const u16* p = base + (size_t)(row0 + (lane & 15)) * ld + (k0 + ((lane >> 4) << 3));
  union { v16bf v; uint4 q[2]; } u;
  u.q[0] = *(const uint4*)(p);
  u.q[1] = *(const uint4*)(p + 16);
  return u.v;
}

__device__ __forceinline__ v8f wmma_bf16(v16bf a, v16bf b, v8f c) {
  return __builtin_amdgcn_wmma_f32_16x16x32_bf16(false, a, false, b, (short)0, c,
                                                 false, false);
}

__device__ __forceinline__ v8f vzero8() {
  v8f z;
#pragma unroll
  for (int i = 0; i < 8; i++) z[i] = 0.0f;
  return z;
}

__device__ __forceinline__ bool win_mask(int wh, int ww, int q, int k) {
  // matches _make_masks with WS=8, SHIFT=4
  bool is_top = (wh == 0), is_left = (ww == 0);
  int qh = q >> 3, qw = q & 7, kh = k >> 3, kw = k & 7;
  bool qa = qh < 4, ka = kh < 4, ql = qw < 4, kl = kw < 4;
  return (is_left && is_top && (ql == kl) && (qa == ka)) ||
         (is_left && !is_top && (ql == kl)) ||
         (!is_left && is_top && (qa == ka)) ||
         (!is_left && !is_top);
}

// ---------------------------------------------------------------- tiny kernels

__global__ void k_f2bf(const float* __restrict__ s, u16* __restrict__ d, int n) {
  int i = blockIdx.x * 256 + threadIdx.x;
  if (i < n) d[i] = f2bf(s[i]);
}

// m[b][j] = sum_c cond[b][c] * w[j][c] + 1
__global__ void k_mod(const float* __restrict__ cond, const float* __restrict__ w,
                      float* __restrict__ m) {
  int b = blockIdx.x, j = threadIdx.x;
  const float* c = cond + b * 256;
  const float* wr = w + j * 256;
  float s = 0.f;
#pragma unroll 8
  for (int k = 0; k < 256; k++) s += c[k] * wr[k];
  m[b * 256 + j] = s + 1.0f;
}

// xn = rmsnorm(x) * m[b]  -> bf16
__global__ void k_rmsnorm(const float* __restrict__ x, const float* __restrict__ m,
                          u16* __restrict__ xn) {
  __shared__ float red[256];
  int row = blockIdx.x, t = threadIdx.x;
  float v = x[(size_t)row * 256 + t];
  red[t] = v * v;
  __syncthreads();
  for (int s = 128; s > 0; s >>= 1) {
    if (t < s) red[t] += red[t + s];
    __syncthreads();
  }
  float inv = rsqrtf(red[0] * (1.0f / 256.0f) + 1e-6f);
  int b = row >> 14;
  xn[(size_t)row * 256 + t] = f2bf(v * inv * m[b * 256 + t]);
}

// ---------------------------------------------------------------- QKV GEMM + qk-norm + RoPE + window scatter

__global__ __launch_bounds__(128)
void k_qkv(const u16* __restrict__ A, const u16* __restrict__ Wb,
           const float* __restrict__ scale_attn, const float* __restrict__ freqs,
           u16* __restrict__ Qb, u16* __restrict__ Kb, u16* __restrict__ Vb) {
  __shared__ float Sc[64 * 68];
  int tid = threadIdx.x, wave = tid >> 5, lane = tid & 31;
  int rowb = blockIdx.x * 64;
  int chunk = blockIdx.y;          // 0..11 : (q|k|v) x head
  int wrow0 = chunk * 64;          // row of qkv_w == output column

  v8f acc[4];
#pragma unroll
  for (int t = 0; t < 4; t++) acc[t] = vzero8();

  for (int k0 = 0; k0 < 256; k0 += 32) {
    if (k0 + 32 < 256)
      __builtin_prefetch(A + (size_t)(rowb + wave * 16 + (lane & 15)) * 256 + k0 + 32, 0, 1);
    v16bf af = load_frag_rm(A, 256, rowb + wave * 16, k0);
#pragma unroll
    for (int t = 0; t < 4; t++) {
      v16bf bf = load_frag_rm(Wb, 256, wrow0 + t * 16, k0);
      acc[t] = wmma_bf16(af, bf, acc[t]);
    }
  }

  int cg = lane & 15, rh = (lane >> 4) << 3;
#pragma unroll
  for (int t = 0; t < 4; t++)
#pragma unroll
    for (int i = 0; i < 8; i++)
      Sc[(wave * 16 + i + rh) * 68 + t * 16 + cg] = acc[t][i];
  __syncthreads();

  // row-wise epilogue: 2 threads per row (32 cols each)
  int r = tid >> 1, half = tid & 1;
  int p = rowb + r;
  int b = p >> 14, hh = (p >> 7) & 127, ww = p & 127;
  int hr = (hh + 4) & 127, wr = (ww + 4) & 127;     // roll(+SHIFT)
  int win = (hr >> 3) * 16 + (wr >> 3);
  int tok = ((hr & 7) << 3) + (wr & 7);
  int mat = chunk >> 2, head = chunk & 3;

  float vals[32];
  float ss = 0.f;
#pragma unroll
  for (int j = 0; j < 32; j++) {
    vals[j] = Sc[r * 68 + half * 32 + j];
    ss += vals[j] * vals[j];
  }
  ss += __shfl_xor(ss, 1);
  size_t base = (((size_t)(b * 4 + head)) * 256 + win) * 64;

  if (mat < 2) {
    float sc = sqrtf(scale_attn[head]) * rsqrtf(ss + 1e-6f);
    if (half == 0) {
      // RoPE on dims 0..31 (pairs (j, j+16)); dims 32..63 untouched (other half)
      float ph_ = -1.f + 2.f * (float)hh * (1.f / 127.f);
      float pw_ = -1.f + 2.f * (float)ww * (1.f / 127.f);
      float o0[16], o1[16];
#pragma unroll
      for (int j = 0; j < 16; j++) {
        float x1 = vals[j] * sc, x2 = vals[j + 16] * sc;
        float th = (j < 8) ? ph_ * freqs[head * 8 + j] : pw_ * freqs[head * 8 + (j - 8)];
        float c = cosf(th), s = sinf(th);
        o0[j] = x1 * c - x2 * s;
        o1[j] = x2 * c + x1 * s;
      }
#pragma unroll
      for (int j = 0; j < 16; j++) { vals[j] = o0[j]; vals[j + 16] = o1[j]; }
    } else {
#pragma unroll
      for (int j = 0; j < 32; j++) vals[j] *= sc;
    }
    u16* dst = (mat == 0 ? Qb : Kb) + (base + tok) * 64 + half * 32;
#pragma unroll
    for (int j = 0; j < 32; j++) dst[j] = f2bf(vals[j]);
  } else {
    // V stored transposed per window: [dim][tok]
    u16* dst = Vb + base * 64 + tok;
#pragma unroll
    for (int j = 0; j < 32; j++) dst[(size_t)(half * 32 + j) * 64] = f2bf(vals[j]);
  }
}

// ---------------------------------------------------------------- windowed attention

__global__ __launch_bounds__(128)
void k_attn(const u16* __restrict__ Qb, const u16* __restrict__ Kb,
            const u16* __restrict__ Vb, u16* __restrict__ Ob) {
  __shared__ u16 Qs[4096], Ks[4096], Vs[4096], Pb[4096];
  __shared__ float Sc[64 * 68];
  int tid = threadIdx.x, wave = tid >> 5, lane = tid & 31;
  int win = blockIdx.x, bh = blockIdx.y;
  int b = bh >> 2, head = bh & 3;
  int wh = win >> 4, wwn = win & 15;
  size_t base = ((size_t)bh * 256 + win) * 4096;

#pragma unroll
  for (int j = 0; j < 4; j++) {
    int off = (tid + 128 * j) * 8;
    *(uint4*)(Qs + off) = *(const uint4*)(Qb + base + off);
    *(uint4*)(Ks + off) = *(const uint4*)(Kb + base + off);
    *(uint4*)(Vs + off) = *(const uint4*)(Vb + base + off);
  }
  __syncthreads();

  // scores = Q K^T  (64x64, K-dim = 64)
  v8f acc[4];
#pragma unroll
  for (int t = 0; t < 4; t++) acc[t] = vzero8();
#pragma unroll
  for (int k0 = 0; k0 < 64; k0 += 32) {
    v16bf af = load_frag_rm(Qs, 64, wave * 16, k0);
#pragma unroll
    for (int t = 0; t < 4; t++)
      acc[t] = wmma_bf16(af, load_frag_rm(Ks, 64, t * 16, k0), acc[t]);
  }
  int cg = lane & 15, rh = (lane >> 4) << 3;
#pragma unroll
  for (int t = 0; t < 4; t++)
#pragma unroll
    for (int i = 0; i < 8; i++)
      Sc[(wave * 16 + i + rh) * 68 + t * 16 + cg] = acc[t][i];
  __syncthreads();

  // masked softmax, 2 threads per row
  int r = tid >> 1, half = tid & 1;
  float vbuf[32];
  float mx = -3e38f;
#pragma unroll
  for (int j = 0; j < 32; j++) {
    int kk = half * 32 + j;
    float v = Sc[r * 68 + kk];
    if (!win_mask(wh, wwn, r, kk)) v = -1e30f;
    vbuf[j] = v;
    mx = fmaxf(mx, v);
  }
  mx = fmaxf(mx, __shfl_xor(mx, 1));
  float sum = 0.f;
#pragma unroll
  for (int j = 0; j < 32; j++) {
    float e = __expf(vbuf[j] - mx);
    vbuf[j] = e;
    sum += e;
  }
  sum += __shfl_xor(sum, 1);
  float invs = 1.0f / sum;
#pragma unroll
  for (int j = 0; j < 32; j++)
    Pb[r * 64 + half * 32 + j] = f2bf(vbuf[j] * invs);
  __syncthreads();

  // O = P V   (V stored [dim][tok] so B-fragment is a row-major load)
  v8f oacc[4];
#pragma unroll
  for (int t = 0; t < 4; t++) oacc[t] = vzero8();
#pragma unroll
  for (int k0 = 0; k0 < 64; k0 += 32) {
    v16bf af = load_frag_rm(Pb, 64, wave * 16, k0);
#pragma unroll
    for (int t = 0; t < 4; t++)
      oacc[t] = wmma_bf16(af, load_frag_rm(Vs, 64, t * 16, k0), oacc[t]);
  }

  // scatter with inverse roll into (pixel, head*64+dim) layout
#pragma unroll
  for (int t = 0; t < 4; t++)
#pragma unroll
    for (int i = 0; i < 8; i++) {
      int q = wave * 16 + i + rh;
      int d = t * 16 + cg;
      int qh = q >> 3, qw = q & 7;
      int hr = wh * 8 + qh, wr2 = wwn * 8 + qw;
      int hh = (hr + 124) & 127, ww2 = (wr2 + 124) & 127;
      size_t rowg = ((size_t)b * 128 + hh) * 128 + ww2;
      Ob[rowg * 256 + head * 64 + d] = f2bf(oacc[t][i]);
    }
}

// ---------------------------------------------------------------- out proj (+skip)

__global__ __launch_bounds__(128)
void k_gemm_out(const u16* __restrict__ A, const u16* __restrict__ Wb,
                const float* __restrict__ skip, float* __restrict__ X2) {
  int tid = threadIdx.x, wave = tid >> 5, lane = tid & 31;
  int rowb = blockIdx.x * 64, colb = blockIdx.y * 64;
  v8f acc[4];
#pragma unroll
  for (int t = 0; t < 4; t++) acc[t] = vzero8();
  for (int k0 = 0; k0 < 256; k0 += 32) {
    if (k0 + 32 < 256)
      __builtin_prefetch(A + (size_t)(rowb + wave * 16 + (lane & 15)) * 256 + k0 + 32, 0, 1);
    v16bf af = load_frag_rm(A, 256, rowb + wave * 16, k0);
#pragma unroll
    for (int t = 0; t < 4; t++)
      acc[t] = wmma_bf16(af, load_frag_rm(Wb, 256, colb + t * 16, k0), acc[t]);
  }
  int cg = lane & 15, rh = (lane >> 4) << 3;
#pragma unroll
  for (int t = 0; t < 4; t++)
#pragma unroll
    for (int i = 0; i < 8; i++) {
      size_t row = rowb + wave * 16 + i + rh;
      int c = colb + t * 16 + cg;
      X2[row * 256 + c] = acc[t][i] + skip[row * 256 + c];
    }
}

// ---------------------------------------------------------------- up proj + GeGLU

__global__ __launch_bounds__(128)
void k_gemm_up(const u16* __restrict__ A, const u16* __restrict__ Wb,
               u16* __restrict__ Hb) {
  int tid = threadIdx.x, wave = tid >> 5, lane = tid & 31;
  int rowb = blockIdx.x * 64, colb = blockIdx.y * 64;   // colb in 0..448 (a-half)
  v8f acca[4], accg[4];
#pragma unroll
  for (int t = 0; t < 4; t++) { acca[t] = vzero8(); accg[t] = vzero8(); }
  for (int k0 = 0; k0 < 256; k0 += 32) {
    v16bf af = load_frag_rm(A, 256, rowb + wave * 16, k0);
#pragma unroll
    for (int t = 0; t < 4; t++) {
      acca[t] = wmma_bf16(af, load_frag_rm(Wb, 256, colb + t * 16, k0), acca[t]);
      accg[t] = wmma_bf16(af, load_frag_rm(Wb, 256, 512 + colb + t * 16, k0), accg[t]);
    }
  }
  int cg = lane & 15, rh = (lane >> 4) << 3;
#pragma unroll
  for (int t = 0; t < 4; t++)
#pragma unroll
    for (int i = 0; i < 8; i++) {
      size_t row = rowb + wave * 16 + i + rh;
      int c = colb + t * 16 + cg;
      float av = acca[t][i], gv = accg[t][i];
      float gel = 0.5f * gv * (1.0f + erff(gv * 0.70710678118654752f));
      Hb[row * 512 + c] = f2bf(av * gel);
    }
}

// ---------------------------------------------------------------- down proj (+skip) -> out

__global__ __launch_bounds__(128)
void k_gemm_down(const u16* __restrict__ A, const u16* __restrict__ Wb,
                 const float* __restrict__ X2, float* __restrict__ out) {
  int tid = threadIdx.x, wave = tid >> 5, lane = tid & 31;
  int rowb = blockIdx.x * 64, colb = blockIdx.y * 64;
  v8f acc[4];
#pragma unroll
  for (int t = 0; t < 4; t++) acc[t] = vzero8();
  for (int k0 = 0; k0 < 512; k0 += 32) {
    if (k0 + 32 < 512)
      __builtin_prefetch(A + (size_t)(rowb + wave * 16 + (lane & 15)) * 512 + k0 + 32, 0, 1);
    v16bf af = load_frag_rm(A, 512, rowb + wave * 16, k0);
#pragma unroll
    for (int t = 0; t < 4; t++)
      acc[t] = wmma_bf16(af, load_frag_rm(Wb, 512, colb + t * 16, k0), acc[t]);
  }
  int cg = lane & 15, rh = (lane >> 4) << 3;
#pragma unroll
  for (int t = 0; t < 4; t++)
#pragma unroll
    for (int i = 0; i < 8; i++) {
      size_t row = rowb + wave * 16 + i + rh;
      int c = colb + t * 16 + cg;
      out[row * 256 + c] = acc[t][i] + X2[row * 256 + c];
    }
}

// ---------------------------------------------------------------- launch

extern "C" void kernel_launch(void* const* d_in, const int* in_sizes, int n_in,
                              void* d_out, int out_size, void* d_ws, size_t ws_size,
                              hipStream_t stream) {
  (void)in_sizes; (void)n_in; (void)out_size; (void)ws_size;
  const float* x          = (const float*)d_in[0];
  const float* cond       = (const float*)d_in[2];
  const float* norm1_w    = (const float*)d_in[3];
  const float* qkv_w      = (const float*)d_in[4];
  const float* scale_attn = (const float*)d_in[5];
  const float* freqs      = (const float*)d_in[6];
  const float* out_w      = (const float*)d_in[7];
  const float* norm2_w    = (const float*)d_in[8];
  const float* up_w       = (const float*)d_in[9];
  const float* down_w     = (const float*)d_in[10];
  float* out = (float*)d_out;

  char* p = (char*)d_ws;
  auto carve = [&](size_t bytes) -> void* {
    void* r = (void*)p;
    p += (bytes + 255) & ~(size_t)255;
    return r;
  };
  u16*   qkv_wb  = (u16*)carve((size_t)768 * 256 * 2);
  u16*   out_wb  = (u16*)carve((size_t)256 * 256 * 2);
  u16*   up_wb   = (u16*)carve((size_t)1024 * 256 * 2);
  u16*   down_wb = (u16*)carve((size_t)256 * 512 * 2);
  float* m1      = (float*)carve(4 * 256 * 4);
  float* m2      = (float*)carve(4 * 256 * 4);
  u16*   XN      = (u16*)carve((size_t)65536 * 256 * 2);
  u16*   Qb      = (u16*)carve((size_t)16777216 * 2);
  u16*   Kb      = (u16*)carve((size_t)16777216 * 2);
  u16*   Vb      = (u16*)carve((size_t)16777216 * 2);
  u16*   Ob      = (u16*)carve((size_t)65536 * 256 * 2);
  float* X2      = (float*)carve((size_t)65536 * 256 * 4);
  u16*   Hb      = Qb;   // reuse Q+K region (64MB) after attention is done

  k_f2bf<<<(768 * 256 + 255) / 256, 256, 0, stream>>>(qkv_w, qkv_wb, 768 * 256);
  k_f2bf<<<(256 * 256 + 255) / 256, 256, 0, stream>>>(out_w, out_wb, 256 * 256);
  k_f2bf<<<(1024 * 256 + 255) / 256, 256, 0, stream>>>(up_w, up_wb, 1024 * 256);
  k_f2bf<<<(256 * 512 + 255) / 256, 256, 0, stream>>>(down_w, down_wb, 256 * 512);
  k_mod<<<4, 256, 0, stream>>>(cond, norm1_w, m1);
  k_mod<<<4, 256, 0, stream>>>(cond, norm2_w, m2);

  k_rmsnorm<<<65536, 256, 0, stream>>>(x, m1, XN);
  k_qkv<<<dim3(1024, 12), 128, 0, stream>>>(XN, qkv_wb, scale_attn, freqs, Qb, Kb, Vb);
  k_attn<<<dim3(256, 16), 128, 0, stream>>>(Qb, Kb, Vb, Ob);
  k_gemm_out<<<dim3(1024, 4), 128, 0, stream>>>(Ob, out_wb, x, X2);
  k_rmsnorm<<<65536, 256, 0, stream>>>(X2, m2, XN);
  k_gemm_up<<<dim3(1024, 8), 128, 0, stream>>>(XN, up_wb, Hb);
  k_gemm_down<<<dim3(1024, 4), 128, 0, stream>>>(Hb, down_wb, X2, out);
}
